// ScatterModule_67774583931141
// MI455X (gfx1250) — matrix operand order repaired and checked by reading, not compiled, not run on hardware
//
#include <hip/hip_runtime.h>

typedef __attribute__((ext_vector_type(16))) __bf16        v16bf;
typedef __attribute__((ext_vector_type(8)))  float         v8f;
typedef __attribute__((ext_vector_type(8)))  unsigned int  v8u;
typedef __attribute__((ext_vector_type(4)))  float         v4f;

#define N_E    1600000
#define DIM    128
#define N_OUT  100000
#define NTILES (N_E / 16)   // 100,000 tiles of 16 rows

__device__ __forceinline__ unsigned fbits(float f) {
    return __builtin_bit_cast(unsigned, f);
}
__device__ __forceinline__ float bfloat(unsigned u) {
    return __builtin_bit_cast(float, u);
}

// Pack 16 floats into bf16-hi and bf16-lo fragments (fp32 = hi + lo split).
__device__ __forceinline__ void split16(const float* fl, v16bf& hi, v16bf& lo) {
    v8u H, L;
#pragma unroll
    for (int i = 0; i < 8; ++i) {
        float x0 = fl[2 * i], x1 = fl[2 * i + 1];
        unsigned h0 = fbits(x0) & 0xFFFF0000u;
        unsigned h1 = fbits(x1) & 0xFFFF0000u;
        H[i] = (h0 >> 16) | h1;
        float r0 = x0 - bfloat(h0);
        float r1 = x1 - bfloat(h1);
        L[i] = (fbits(r0) >> 16) | (fbits(r1) & 0xFFFF0000u);
    }
    hi = __builtin_bit_cast(v16bf, H);
    lo = __builtin_bit_cast(v16bf, L);
}

__global__ void zero_out_kernel(float* __restrict__ out, int n) {
    int n4 = n >> 2;
    v4f z = {0.f, 0.f, 0.f, 0.f};
    for (int i = blockIdx.x * blockDim.x + threadIdx.x; i < n4;
         i += gridDim.x * blockDim.x) {
        *(v4f*)(out + 4 * i) = z;
    }
    // tail (n is a multiple of 4 here, but stay safe)
    int tid = blockIdx.x * blockDim.x + threadIdx.x;
    if (tid < (n & 3)) out[(n & ~3) + tid] = 0.f;
}

__global__ void __launch_bounds__(256)
fused_linear_relu_scatter(const float* __restrict__ e,
                          const int* __restrict__ index,
                          const float* __restrict__ W,
                          const float* __restrict__ bias,
                          float* __restrict__ out) {
    // W split into bf16 hi/lo, row-padded to 136 elements (272B) so the 16
    // lane-rows of a B-fragment land on distinct LDS banks.
    __shared__ __attribute__((aligned(16))) __bf16 Whi[128][136];
    __shared__ __attribute__((aligned(16))) __bf16 Wlo[128][136];
    __shared__ float Bias[128];

    const int t = threadIdx.x;

    // ---- one-time W conversion into LDS (64KB fp32 -> 2x bf16 planes) ----
    for (int i = t; i < 128 * 128; i += 256) {
        int r = i >> 7, c = i & 127;
        float w = W[i];
        unsigned hu = fbits(w) & 0xFFFF0000u;
        float rem = w - bfloat(hu);
        Whi[r][c] = __builtin_bit_cast(__bf16, (unsigned short)(hu >> 16));
        Wlo[r][c] = __builtin_bit_cast(__bf16, (unsigned short)(fbits(rem) >> 16));
    }
    if (t < 128) Bias[t] = bias[t];
    __syncthreads();

    const int lane    = t & 31;
    const int col     = lane & 15;   // A row within tile / feature within n-tile
    const int halfSel = lane >> 4;   // lane half selects 16-wide K half / row+8
    const int waveId  = (blockIdx.x << 3) | (t >> 5);
    const int nWaves  = gridDim.x << 3;

    for (int tile = waveId; tile < NTILES; tile += nWaves) {
        const float* rp = e + (long)(tile * 16 + col) * DIM;

        // Prefetch next tile's row while we compute this one.
        if (tile + nWaves < NTILES) {
            __builtin_prefetch(e + (long)((tile + nWaves) * 16 + col) * DIM, 0, 0);
        }

        // ---- build A fragments: 16x32 A, contiguous k order (k = 16*half + s) ----
        // lanes 0-15 hold K = 0..15 +ks*32; lanes 16-31 hold 16..31 +ks*32
        const int aoff = halfSel * 16;
        v16bf a_hi[4], a_lo[4];
#pragma unroll
        for (int ks = 0; ks < 4; ++ks) {
            const int k0 = ks * 32 + aoff;
            float fl[16];
            v4f x0 = *(const v4f*)(rp + k0);
            v4f x1 = *(const v4f*)(rp + k0 + 4);
            v4f x2 = *(const v4f*)(rp + k0 + 8);
            v4f x3 = *(const v4f*)(rp + k0 + 12);
#pragma unroll
            for (int j = 0; j < 4; ++j) {
                fl[j]      = x0[j];
                fl[4 + j]  = x1[j];
                fl[8 + j]  = x2[j];
                fl[12 + j] = x3[j];
            }
            split16(fl, a_hi[ks], a_lo[ks]);
        }

        // ---- segment ids for this lane's 8 output rows (r or r+8) ----
        const int segBase = tile * 16 + halfSel * 8;
        int seg[8];
#pragma unroll
        for (int r = 0; r < 8; ++r) seg[r] = index[segBase + r];

        // ---- GEMM over 8 feature tiles; bf16x3 fp32 emulation ----
#pragma unroll
        for (int nt = 0; nt < 8; ++nt) {
            v8f acc = {0.f, 0.f, 0.f, 0.f, 0.f, 0.f, 0.f, 0.f};
            const int featRow = nt * 16 + col;
#pragma unroll
            for (int ks = 0; ks < 4; ++ks) {
                // B 32x16 layout: lane col = feature, lanes 0-15 K=0..15,
                // lanes 16-31 K=16..31 (within this ks*32 chunk)
                const int kofs = ks * 32 + halfSel * 16;
                v16bf bh = __builtin_bit_cast(v16bf, *(const v8u*)&Whi[featRow][kofs]);
                v16bf bl = __builtin_bit_cast(v16bf, *(const v8u*)&Wlo[featRow][kofs]);
                acc = __builtin_amdgcn_wmma_f32_16x16x32_bf16(
                    false, a_hi[ks], false, bh, (short)0, acc, false, false);
                acc = __builtin_amdgcn_wmma_f32_16x16x32_bf16(
                    false, a_lo[ks], false, bh, (short)0, acc, false, false);
                acc = __builtin_amdgcn_wmma_f32_16x16x32_bf16(
                    false, a_hi[ks], false, bl, (short)0, acc, false, false);
            }
            // ---- bias + ReLU + scatter (L2-resident fp32 atomics) ----
            const float bv = Bias[featRow];
#pragma unroll
            for (int r = 0; r < 8; ++r) {
                float v = acc[r] + bv;
                v = v > 0.f ? v : 0.f;
                __hip_atomic_fetch_add(out + (long)seg[r] * DIM + featRow, v,
                                       __ATOMIC_RELAXED, __HIP_MEMORY_SCOPE_AGENT);
            }
        }
    }
}

extern "C" void kernel_launch(void* const* d_in, const int* in_sizes, int n_in,
                              void* d_out, int out_size, void* d_ws, size_t ws_size,
                              hipStream_t stream) {
    const float* e     = (const float*)d_in[0];
    const int*   index = (const int*)d_in[1];
    const float* W     = (const float*)d_in[2];
    const float* bias  = (const float*)d_in[3];
    float*       out   = (float*)d_out;

    // Atomic accumulation needs a zeroed output every call (deterministic).
    zero_out_kernel<<<4096, 256, 0, stream>>>(out, out_size);

    // 1250 WGs x 8 waves = 10,000 waves -> exactly 10 row-tiles per wave.
    fused_linear_relu_scatter<<<1250, 256, 0, stream>>>(e, index, W, bias, out);
}
